// MPNN_19164144074848
// MI455X (gfx1250) — compile-verified, matrix operand order checked by reading, not run on hardware
//
#include <hip/hip_runtime.h>
#include <math.h>

#define NATOMS 20000
#define NEDGES 640000
#define NWAVE  8
#define NANG   9
#define NORB   32
#define ITERS  3

typedef __attribute__((ext_vector_type(16))) _Float16 v16h;
typedef __attribute__((ext_vector_type(8)))  float    v8f;

#if __has_builtin(__builtin_amdgcn_rcpf)
#define FAST_RCP(x) __builtin_amdgcn_rcpf(x)
#else
#define FAST_RCP(x) (1.f / (x))
#endif
#if __has_builtin(__builtin_amdgcn_rsqf)
#define FAST_RSQ(x) __builtin_amdgcn_rsqf(x)
#else
#define FAST_RSQ(x) rsqrtf(x)
#endif

__device__ __forceinline__ float silu_f(float x) {
  return x * FAST_RCP(1.f + __expf(-x));
}
__device__ __forceinline__ void atom_add_f32(float* p, float v) {
  __hip_atomic_fetch_add(p, v, __ATOMIC_RELAXED, __HIP_MEMORY_SCOPE_AGENT);
}

// ---------------- utility kernels ----------------
__global__ void fill_zero_kernel(float* p, int n) {
  for (int i = blockIdx.x * blockDim.x + threadIdx.x; i < n; i += gridDim.x * blockDim.x)
    p[i] = 0.f;
}
__global__ void copy_kernel(float* __restrict__ dst, const float* __restrict__ src, int n) {
  for (int i = blockIdx.x * blockDim.x + threadIdx.x; i < n; i += gridDim.x * blockDim.x)
    dst[i] = src[i];
}
__global__ void cvt_f16_kernel(const float* __restrict__ src, _Float16* __restrict__ dst, int n) {
  for (int i = blockIdx.x * blockDim.x + threadIdx.x; i < n; i += gridDim.x * blockDim.x)
    dst[i] = (_Float16)src[i];
}

// ---------------- embedding MLP (1 -> 8 -> 8 -> 24), per-atom scalar ----------------
__device__ __forceinline__ void ln8(float* h) {
  float mu = 0.f;
  #pragma unroll
  for (int j = 0; j < 8; ++j) mu += h[j];
  mu *= 0.125f;
  float var = 0.f;
  #pragma unroll
  for (int j = 0; j < 8; ++j) { float d = h[j] - mu; var += d * d; }
  var *= 0.125f;
  float inv = FAST_RSQ(var + 1e-5f);
  #pragma unroll
  for (int j = 0; j < 8; ++j) h[j] = (h[j] - mu) * inv;
}

__global__ void emb_kernel(const float* __restrict__ species,
                           const float* __restrict__ W0, const float* __restrict__ b0,
                           const float* __restrict__ W1, const float* __restrict__ b1,
                           const float* __restrict__ W2, const float* __restrict__ b2,
                           float* __restrict__ center_coeff) {
  int a = blockIdx.x * blockDim.x + threadIdx.x;
  if (a >= NATOMS) return;
  float x = species[a];
  float h[8], h2[8];
  #pragma unroll
  for (int j = 0; j < 8; ++j) h[j] = silu_f(x * W0[j] + b0[j]);
  ln8(h);
  #pragma unroll
  for (int j = 0; j < 8; ++j) {
    float s = b1[j];
    #pragma unroll
    for (int i = 0; i < 8; ++i) s = fmaf(h[i], W1[i * 8 + j], s);
    h2[j] = silu_f(s);
  }
  ln8(h2);
  #pragma unroll
  for (int j = 0; j < 24; ++j) {
    float s = b2[j];
    #pragma unroll
    for (int i = 0; i < 8; ++i) s = fmaf(h2[i], W2[i * 24 + j], s);
    center_coeff[a * 24 + j] = s;
  }
}

// ---------------- initial edge kernel: radial/sph/cut + scatter orbital ----------------
__global__ void edge_init_kernel(const float* __restrict__ cart, const int* __restrict__ nl,
                                 const float* __restrict__ shifts, const float* __restrict__ nfac,
                                 const float* __restrict__ center_coeff,
                                 float* __restrict__ rc, float* __restrict__ sphb,
                                 float* __restrict__ cutb, float* __restrict__ co) {
  int e = blockIdx.x * blockDim.x + threadIdx.x;
  if (e >= NEDGES) return;
  int c = nl[e], n = nl[NEDGES + e];
  float dx = cart[3 * n + 0] - cart[3 * c + 0] + shifts[3 * e + 0];
  float dy = cart[3 * n + 1] - cart[3 * c + 1] + shifts[3 * e + 1];
  float dz = cart[3 * n + 2] - cart[3 * c + 2] + shifts[3 * e + 2];
  float d2 = dx * dx + dy * dy + dz * dz;
  float d = sqrtf(d2);
  float rinv = FAST_RSQ(d2 + 1e-12f);
  float x = dx * rinv, y = dy * rinv, z = dz * rinv;
  float t = 0.5f * __cosf(d * 0.78539816339744831f) + 0.5f; // pi/CUTOFF
  float cut = nfac[e] * t * t;

  const float* cc_c = center_coeff + 24 * c;
  const float* cc_n = center_coeff + 24 * n;
  float rcl[8];
  #pragma unroll
  for (int j = 0; j < 8; ++j) {
    float coef = cc_c[j] * cc_n[j];
    float rs   = cc_c[8 + j] * cc_n[8 + j];
    float al   = cc_c[16 + j] * cc_n[16 + j];
    float tt = al * (d - rs);
    rcl[j] = cut * __expf(-tt * tt) * coef;
    rc[e * 8 + j] = rcl[j];
  }
  float sp[9];
  sp[0] = 0.28209479177387814f;
  sp[1] = 0.4886025119029199f * y;
  sp[2] = 0.4886025119029199f * z;
  sp[3] = 0.4886025119029199f * x;
  sp[4] = 1.0925484305920792f * x * y;
  sp[5] = 1.0925484305920792f * y * z;
  sp[6] = 0.31539156525252005f * (3.f * z * z - 1.f);
  sp[7] = 1.0925484305920792f * x * z;
  sp[8] = 0.5462742152960396f * (x * x - y * y);
  #pragma unroll
  for (int k = 0; k < 9; ++k) sphb[e * 9 + k] = sp[k];
  cutb[e] = cut;

  float* cobase = co + 72 * c;
  #pragma unroll
  for (int k = 0; k < 9; ++k)
    #pragma unroll
    for (int j = 0; j < 8; ++j)
      atom_add_f32(&cobase[k * 8 + j], rcl[j] * sp[k]);
}

// ---------------- iteration edge kernel ----------------
__global__ void edge_iter_kernel(const int* __restrict__ nl, const float* __restrict__ rc,
                                 const float* __restrict__ sphb, const float* __restrict__ cutb,
                                 const float* __restrict__ iter_coeff,
                                 const float* __restrict__ co_old, float* __restrict__ co_new) {
  int e = blockIdx.x * blockDim.x + threadIdx.x;
  if (e >= NEDGES) return;
  int c = nl[e], n = nl[NEDGES + e];
  float cut = cutb[e];
  float icf[8];
  #pragma unroll
  for (int j = 0; j < 8; ++j) icf[j] = iter_coeff[n * 8 + j] * rc[e * 8 + j];
  const float* con = co_old + 72 * n;
  float*       coc = co_new + 72 * c;
  #pragma unroll
  for (int k = 0; k < 9; ++k) {
    float sk = sphb[e * 9 + k];
    #pragma unroll
    for (int j = 0; j < 8; ++j) {
      float w = fmaf(icf[j], sk, con[k * 8 + j] * cut);
      atom_add_f32(&coc[k * 8 + j], w);
    }
  }
}

// ---------------- contraction: density[a,m] += sum_k (sum_j co[a,k,j]*cc[j,m])^2 ----------------
__global__ void contract_kernel(const float* __restrict__ co, const float* __restrict__ cc,
                                float* __restrict__ density) {
  int wave = threadIdx.x >> 5;
  int lane = threadIdx.x & 31;
  int a = blockIdx.x * 4 + wave;
  if (a >= NATOMS) return;
  const float* coa = co + 72 * a;
  float dsum = 0.f;
  #pragma unroll
  for (int k = 0; k < 9; ++k) {
    float s = 0.f;
    #pragma unroll
    for (int j = 0; j < 8; ++j) s = fmaf(coa[k * 8 + j], cc[j * 32 + lane], s);
    dsum = fmaf(s, s, dsum);
  }
  density[a * 32 + lane] += dsum;
}

// ---------------- WMMA helper: 16x16 f32 tile of A(16xK f16) @ W(KxN f16) ----------------
template <int K>
__device__ __forceinline__ v8f wmma_frag(const _Float16* __restrict__ A, int lda,
                                         const _Float16* __restrict__ W, int ldw,
                                         int colBase, int ncols, v8f acc) {
  int lane = threadIdx.x & 31;
  int m    = lane & 15;
  int kh   = (lane >> 4) & 1;          // which K-half this lane holds (ISA 16-bit A layout)
  int col  = colBase + m;
  bool cok = (col < ncols);
  #pragma unroll
  for (int k0 = 0; k0 < K; k0 += 32) {
    v16h a, b;
    #pragma unroll
    for (int i = 0; i < 8; ++i) {
      int kA = k0 + kh * 8 + i;        // K in {0..7} / {8..15}
      int kB = k0 + 16 + kh * 8 + i;   // K in {16..23} / {24..31}
      a[i]     = A[m * lda + kA];
      a[i + 8] = A[m * lda + kB];
      b[i]     = cok ? W[kA * ldw + col] : (_Float16)0.f;
      b[i + 8] = cok ? W[kB * ldw + col] : (_Float16)0.f;
    }
    acc = __builtin_amdgcn_wmma_f32_16x16x32_f16(false, a, false, b, (short)0, acc,
                                                 false, false);
  }
  return acc;
}

// parallel layernorm: 8 lanes per row, shuffle-reduce, write f16
__device__ __forceinline__ void ln_rows(const float (*Sf)[64], _Float16 (*Ah)[72]) {
  int tid = threadIdx.x;
  int row = tid >> 3;   // 0..15
  int sub = tid & 7;    // 0..7 (aligned 8-lane groups inside a wave32)
  float v[8];
  float s = 0.f;
  #pragma unroll
  for (int j = 0; j < 8; ++j) { v[j] = Sf[row][sub * 8 + j]; s += v[j]; }
  #pragma unroll
  for (int m = 4; m; m >>= 1) s += __shfl_xor(s, m, 8);
  float mu = s * (1.f / 64.f);
  float s2 = 0.f;
  #pragma unroll
  for (int j = 0; j < 8; ++j) { float d = v[j] - mu; s2 += d * d; }
  #pragma unroll
  for (int m = 4; m; m >>= 1) s2 += __shfl_xor(s2, m, 8);
  float inv = FAST_RSQ(s2 * (1.f / 64.f) + 1e-5f);
  #pragma unroll
  for (int j = 0; j < 8; ++j) Ah[row][sub * 8 + j] = (_Float16)((v[j] - mu) * inv);
}

// ---------------- MLP (32 -> 64 -> 64 -> out_dim) with silu+layernorm, WMMA tiles ----------------
__launch_bounds__(128)
__global__ void mlp_wmma_kernel(const float* __restrict__ X, int n_rows,
                                const _Float16* __restrict__ W0h, const float* __restrict__ b0,
                                const _Float16* __restrict__ W1h, const float* __restrict__ b1,
                                const _Float16* __restrict__ W2h, const float* __restrict__ b2,
                                float* __restrict__ Y, int out_dim) {
  __shared__ _Float16 Ah[16][72];   // f16 activations (K up to 64, padded)
  __shared__ float    Sf[16][64];   // f32 pre-layernorm activations
  const int tid  = threadIdx.x;
  const int lane = tid & 31;
  const int wave = tid >> 5;
  const int rowBase = blockIdx.x * 16;

  // stage input tile (16 x 32) as f16
  for (int i = tid; i < 16 * 32; i += 128) {
    int r = i >> 5, ccol = i & 31;
    int gr = rowBase + r;
    Ah[r][ccol] = (_Float16)((gr < n_rows) ? X[gr * 32 + ccol] : 0.f);
  }
  __syncthreads();

  const int col = (lane & 15) + wave * 16;
  const int rhi = (lane >> 4) * 8;

  // layer 0: K=32, N=64 (wave w -> 16-col strip)
  {
    v8f acc = {};
    acc = wmma_frag<32>(&Ah[0][0], 72, W0h, 64, wave * 16, 64, acc);
    float bv = b0[col];
    #pragma unroll
    for (int r = 0; r < 8; ++r) Sf[r + rhi][col] = silu_f(acc[r] + bv);
  }
  __syncthreads();
  ln_rows(Sf, Ah);
  __syncthreads();

  // layer 1: K=64, N=64
  {
    v8f acc = {};
    acc = wmma_frag<64>(&Ah[0][0], 72, W1h, 64, wave * 16, 64, acc);
    float bv = b1[col];
    #pragma unroll
    for (int r = 0; r < 8; ++r) Sf[r + rhi][col] = silu_f(acc[r] + bv);
  }
  __syncthreads();
  ln_rows(Sf, Ah);
  __syncthreads();

  // layer 2: K=64, N=out_dim (<=16) -> wave 0 only, no activation
  if (wave == 0) {
    v8f acc = {};
    acc = wmma_frag<64>(&Ah[0][0], 72, W2h, out_dim, 0, out_dim, acc);
    int oc = lane & 15;
    if (oc < out_dim) {
      float bv = b2[oc];
      #pragma unroll
      for (int r = 0; r < 8; ++r) {
        int row = rowBase + r + rhi;
        if (row < n_rows) Y[row * out_dim + oc] = acc[r] + bv;
      }
    }
  }
}

// ---------------- weighted final reduction ----------------
__global__ void reduce_kernel(const float* __restrict__ ov, const float* __restrict__ cf,
                              int n, float* __restrict__ out) {
  __shared__ float red[256];
  int tid = threadIdx.x;
  float s = 0.f;
  for (int i = blockIdx.x * blockDim.x + tid; i < n; i += gridDim.x * blockDim.x)
    s += ov[i] * cf[i];
  red[tid] = s;
  __syncthreads();
  for (int off = 128; off > 0; off >>= 1) {
    if (tid < off) red[tid] += red[tid + off];
    __syncthreads();
  }
  if (tid == 0) atom_add_f32(out, red[0]);
}

// ---------------- host orchestration ----------------
extern "C" void kernel_launch(void* const* d_in, const int* in_sizes, int n_in,
                              void* d_out, int out_size, void* d_ws, size_t ws_size,
                              hipStream_t stream) {
  (void)in_sizes; (void)n_in; (void)ws_size; (void)out_size;
  const float* cart    = (const float*)d_in[0];
  const int*   nl      = (const int*)  d_in[1];
  const float* shifts  = (const float*)d_in[2];
  const float* cfac    = (const float*)d_in[3];
  const float* nfac    = (const float*)d_in[4];
  const float* species = (const float*)d_in[5];
  const float* cc      = (const float*)d_in[6];
  const float* eW0 = (const float*)d_in[7],  *eb0 = (const float*)d_in[8];
  const float* eW1 = (const float*)d_in[9],  *eb1 = (const float*)d_in[10];
  const float* eW2 = (const float*)d_in[11], *eb2 = (const float*)d_in[12];
  float* out = (float*)d_out;

  float* ws = (float*)d_ws;
  size_t off = 0;
  float* coA     = ws + off; off += (size_t)NATOMS * 72;
  float* coB     = ws + off; off += (size_t)NATOMS * 72;
  float* density = ws + off; off += (size_t)NATOMS * 32;
  float* ccoeff  = ws + off; off += (size_t)NATOMS * 24;
  float* rc      = ws + off; off += (size_t)NEDGES * 8;
  float* sphb    = ws + off; off += (size_t)NEDGES * 9;
  float* cutb    = ws + off; off += (size_t)NEDGES;
  float* itc     = ws + off; off += (size_t)NATOMS * 8;
  float* outv    = ws + off; off += (size_t)NATOMS;
  _Float16* wh   = (_Float16*)(ws + off);   // f16 weight staging

  // pre-convert MLP weights to f16 (layouts preserved)
  // per-MLP: W0 32*64=2048, W1 64*64=4096, W2 64*out
  _Float16* w16[4][3];
  {
    size_t hoff = 0;
    for (int m = 0; m < 4; ++m) {
      int base = (m < 3) ? (13 + 6 * m) : 31;
      int odim = (m < 3) ? 8 : 1;
      int sizes[3] = { 32 * 64, 64 * 64, 64 * odim };
      for (int wgt = 0; wgt < 3; ++wgt) {
        w16[m][wgt] = wh + hoff;
        cvt_f16_kernel<<<16, 256, 0, stream>>>((const float*)d_in[base + 2 * wgt],
                                               w16[m][wgt], sizes[wgt]);
        hoff += (size_t)sizes[wgt];
      }
    }
  }

  fill_zero_kernel<<<512, 256, 0, stream>>>(coA, NATOMS * 72);
  fill_zero_kernel<<<512, 256, 0, stream>>>(density, NATOMS * 32);
  fill_zero_kernel<<<1, 32, 0, stream>>>(out, 1);

  emb_kernel<<<(NATOMS + 255) / 256, 256, 0, stream>>>(species, eW0, eb0, eW1, eb1, eW2, eb2, ccoeff);
  edge_init_kernel<<<(NEDGES + 255) / 256, 256, 0, stream>>>(cart, nl, shifts, nfac, ccoeff,
                                                             rc, sphb, cutb, coA);
  contract_kernel<<<(NATOMS + 3) / 4, 128, 0, stream>>>(coA, cc, density);

  float* co_cur = coA;
  float* co_nxt = coB;
  for (int l = 0; l < ITERS; ++l) {
    const float* b0 = (const float*)d_in[14 + 6 * l];
    const float* b1 = (const float*)d_in[16 + 6 * l];
    const float* b2 = (const float*)d_in[18 + 6 * l];
    mlp_wmma_kernel<<<(NATOMS + 15) / 16, 128, 0, stream>>>(density, NATOMS,
                                                            w16[l][0], b0, w16[l][1], b1,
                                                            w16[l][2], b2, itc, 8);
    copy_kernel<<<512, 256, 0, stream>>>(co_nxt, co_cur, NATOMS * 72);
    edge_iter_kernel<<<(NEDGES + 255) / 256, 256, 0, stream>>>(nl, rc, sphb, cutb, itc,
                                                               co_cur, co_nxt);
    float* t = co_cur; co_cur = co_nxt; co_nxt = t;
    contract_kernel<<<(NATOMS + 3) / 4, 128, 0, stream>>>(co_cur, cc, density);
  }

  const float* ob0 = (const float*)d_in[32];
  const float* ob1 = (const float*)d_in[34];
  const float* ob2 = (const float*)d_in[36];
  mlp_wmma_kernel<<<(NATOMS + 15) / 16, 128, 0, stream>>>(density, NATOMS,
                                                          w16[3][0], ob0, w16[3][1], ob1,
                                                          w16[3][2], ob2, outv, 1);
  reduce_kernel<<<256, 256, 0, stream>>>(outv, cfac, NATOMS, out);
}
